// ScaledDotProductAttention_17428977287939
// MI455X (gfx1250) — matrix operand order repaired and checked by reading, not compiled, not run on hardware
//
#include <hip/hip_runtime.h>

// ---------------------------------------------------------------------------
// Flash attention for B=4,H=8,S=2048,DK=64 (fp32 in/out) on gfx1250 (CDNA5).
//  Pass 1: K f32 -> bf16 (same layout);  V f32 -> bf16 transposed [d][s].
//  Pass 2: flash attention, bf16 WMMA 16x16x32 f32-accum, online softmax,
//          K/V tiles DMA'd into LDS by the Tensor Data Mover (double buffered,
//          TDM pad feature reproduces the 144B padded row pitch), DPP16
//          butterfly reductions for the softmax row stats.
// ---------------------------------------------------------------------------

#define B_    4
#define H_    8
#define S_    2048
#define DK_   64
#define BH_   (B_ * H_)
#define QTILE 128
#define KTILE 64
#define PITCH 72            // padded bf16 row pitch (144B)
#define NT    (S_ / KTILE)  // 32 key tiles

typedef __attribute__((ext_vector_type(16))) __bf16 v16bf;
typedef __attribute__((ext_vector_type(8)))  __bf16 v8bf;
typedef __attribute__((ext_vector_type(4)))  __bf16 v4bf;
typedef __attribute__((ext_vector_type(8)))  float  v8f;
typedef __attribute__((ext_vector_type(4)))  float  v4f;
typedef __attribute__((ext_vector_type(4)))  unsigned int v4u;
typedef __attribute__((ext_vector_type(8)))  int    v8i;
typedef __attribute__((ext_vector_type(4)))  int    v4i;

// ---- DPP16 butterfly reduction within each 16-lane row (wave32) -------------
template <int CTRL>
static __device__ __forceinline__ float dpp_f(float v) {
  return __int_as_float(
      __builtin_amdgcn_mov_dpp(__float_as_int(v), CTRL, 0xF, 0xF, true));
}
static __device__ __forceinline__ float redMax16(float v) {
  v = fmaxf(v, dpp_f<0xB1>(v));   // quad_perm(1,0,3,2)  : xor 1
  v = fmaxf(v, dpp_f<0x4E>(v));   // quad_perm(2,3,0,1)  : xor 2
  v = fmaxf(v, dpp_f<0x141>(v));  // row_half_mirror     : xor 4 (quads uniform)
  v = fmaxf(v, dpp_f<0x140>(v));  // row_mirror          : xor 8
  return v;
}
static __device__ __forceinline__ float redSum16(float v) {
  v += dpp_f<0xB1>(v);
  v += dpp_f<0x4E>(v);
  v += dpp_f<0x141>(v);
  v += dpp_f<0x140>(v);
  return v;
}

// ---- WMMA fragment gathers (layouts per CDNA5 ISA 7.12.2) -------------------
// A (16x32 bf16): lane = m + 16*hi; k-chunks at hi*8 and 16+hi*8.
static __device__ __forceinline__ v16bf loadAfrag(const __bf16* rowp, int hi) {
  union { v16bf v; v8bf h[2]; } u;
  u.h[0] = *(const v8bf*)(rowp + hi * 8);
  u.h[1] = *(const v8bf*)(rowp + 16 + hi * 8);
  return u.v;
}
// B (32x16 bf16): lane = n + 16*hi; k-chunks at hi*8 and 16+hi*8.
static __device__ __forceinline__ v16bf loadBfrag(const __bf16* rowp, int hi) {
  union { v16bf v; v8bf h[2]; } u;
  u.h[0] = *(const v8bf*)(rowp + hi * 8);
  u.h[1] = *(const v8bf*)(rowp + 16 + hi * 8);
  return u.v;
}

// ---- Tensor Data Mover: 2D bf16 tile -> LDS with pad (PITCH rows) -----------
// group0/group1 packing per CDNA5 ISA 8.3/8.4. tile interior => tensor dims =
// tile dims; global_addr = tile start. pad: every 32 DWORDs insert 4 DWORDs.
static __device__ __forceinline__ void tdm_load_tile_2d(
    unsigned int lds_off, const void* gptr,
    unsigned int tile_d0, unsigned int tile_d1, unsigned long long stride0) {
  v4u g0;
  unsigned long long ga = (unsigned long long)gptr;
  g0[0] = 1u;                                        // count=1 (valid), user desc
  g0[1] = lds_off;                                   // lds_addr (bytes)
  g0[2] = (unsigned int)(ga & 0xFFFFFFFFu);          // global_addr[31:0]
  g0[3] = (unsigned int)((ga >> 32) & 0x01FFFFFFu)   // global_addr[56:32]
        | (2u << 30);                                // type = 2 ("image")

  const unsigned int td0 = tile_d0, td1 = tile_d1;   // tensor dims = tile dims
  v8i g1;
  g1[0] = (int)((1u << 16)        // data_size = 2 bytes
              | (1u << 20)        // pad_enable
              | (4u << 22)        // pad_interval: 32 DWORDs (128B)
              | (3u << 25));      // pad_amount:   4 DWORDs (16B)
  g1[1] = (int)((td0 & 0xFFFFu) << 16);                               // tensor_dim0[15:0]
  g1[2] = (int)(((td0 >> 16) & 0xFFFFu) | ((td1 & 0xFFFFu) << 16));   // dim0[31:16] | dim1[15:0]
  g1[3] = (int)(((td1 >> 16) & 0xFFFFu) | ((tile_d0 & 0xFFFFu) << 16)); // dim1[31:16] | tile_dim0
  g1[4] = (int)(tile_d1 & 0xFFFFu);                                   // tile_dim1; tile_dim2=0
  g1[5] = (int)(unsigned int)(stride0 & 0xFFFFFFFFull);               // dim0_stride[31:0]
  g1[6] = (int)(unsigned int)((stride0 >> 32) & 0xFFFFull);           // dim0_stride[47:32]
  g1[7] = 0;                                                          // dim1_stride unused (2D)

  v4i z4 = {0, 0, 0, 0};
#if __clang_major__ >= 23
  v8i z8 = {0, 0, 0, 0, 0, 0, 0, 0};
  __builtin_amdgcn_tensor_load_to_lds(g0, g1, z4, z4, z8, 0);
#else
  __builtin_amdgcn_tensor_load_to_lds(g0, g1, z4, z4, 0);
#endif
}

static __device__ __forceinline__ unsigned int lds_offset(const void* p) {
  // generic pointers into LDS carry the LDS byte offset in their low 32 bits
  return (unsigned int)(unsigned long long)(uintptr_t)p;
}

// ============================ pre-pass kernels ==============================

__global__ __launch_bounds__(256)
void convert_k_kernel(const float* __restrict__ in, __bf16* __restrict__ out, int n4) {
  int i = blockIdx.x * blockDim.x + threadIdx.x;
  if (i < n4) {
    v4f f = *(const v4f*)(in + (size_t)i * 4);
    v4bf h = { (__bf16)f.x, (__bf16)f.y, (__bf16)f.z, (__bf16)f.w };
    *(v4bf*)(out + (size_t)i * 4) = h;
  }
}

__global__ __launch_bounds__(256)
void transpose_v_kernel(const float* __restrict__ V, __bf16* __restrict__ Vt) {
  __shared__ __align__(16) __bf16 sT[DK_ * PITCH];   // [feature][key] 64x64 (+pad)
  const int bh = blockIdx.y;
  const int s0 = blockIdx.x * 64;
  const int tid = threadIdx.x;
  const float* Vp = V + (size_t)bh * S_ * DK_;

  const int c4 = tid & 15;
  #pragma unroll
  for (int p = 0; p < 4; ++p) {
    int row = p * 16 + (tid >> 4);                   // key within tile
    v4f f = *(const v4f*)(Vp + (size_t)(s0 + row) * DK_ + c4 * 4);
    sT[(c4 * 4 + 0) * PITCH + row] = (__bf16)f.x;
    sT[(c4 * 4 + 1) * PITCH + row] = (__bf16)f.y;
    sT[(c4 * 4 + 2) * PITCH + row] = (__bf16)f.z;
    sT[(c4 * 4 + 3) * PITCH + row] = (__bf16)f.w;
  }
  __syncthreads();

  __bf16* Op = Vt + (size_t)bh * DK_ * S_;
  const int f  = tid >> 2;                            // feature 0..63
  const int k0 = (tid & 3) * 16;                      // key chunk
  v8bf a = *(const v8bf*)(sT + f * PITCH + k0);
  v8bf b = *(const v8bf*)(sT + f * PITCH + k0 + 8);
  *(v8bf*)(Op + (size_t)f * S_ + s0 + k0)     = a;   // coalesced 32B stores
  *(v8bf*)(Op + (size_t)f * S_ + s0 + k0 + 8) = b;
}

// ============================ main kernel ===================================

__global__ __launch_bounds__(256)
void flash_attn_cdna5_kernel(const float* __restrict__ Q,
                             const __bf16* __restrict__ Kb,   // [bh][s][d] bf16
                             const __bf16* __restrict__ Vtb,  // [bh][d][s] bf16
                             const int*   __restrict__ MaskP,
                             float*       __restrict__ O) {
  __shared__ __align__(16) __bf16 sQ[QTILE * PITCH];       // per-wave slices; reused as P
  __shared__ __align__(16) __bf16 sK[2][KTILE * PITCH];    // [key][feature] (TDM-filled)
  __shared__ __align__(16) __bf16 sVt[2][DK_ * PITCH];     // [feature][key] (TDM-filled)

  const int tid  = threadIdx.x;
  const int lane = tid & 31;
  const int wave = tid >> 5;
  const int hi   = lane >> 4;
  const int lm   = lane & 15;
  const int bh   = blockIdx.y;
  const bool masked = (MaskP[bh] == 1);

  const float* Qp = Q + (size_t)bh * S_ * DK_;
  const __bf16* Kp = Kb + (size_t)bh * S_ * DK_;
  const __bf16* Vp = Vtb + (size_t)bh * DK_ * S_;
  float* Op = O + (size_t)bh * S_ * DK_;

  const int q0 = blockIdx.x * QTILE + wave * 16;
  __bf16* sQw = sQ + wave * 16 * PITCH;

  // ---- TDM prologue: DMA tile 0 into buffer 0 (one wave issues) ----
  if (wave == 0) {
    tdm_load_tile_2d(lds_offset(&sK[0][0]),  Kp, 64, 64, (unsigned long long)DK_);
    tdm_load_tile_2d(lds_offset(&sVt[0][0]), Vp, 64, 64, (unsigned long long)S_);
  }

  // ---- stage this wave's 16x64 Q slice to LDS as bf16 ----
  #pragma unroll
  for (int i = 0; i < 8; ++i) {
    int s  = i * 32 + lane;
    int r  = s >> 4;
    int c4 = s & 15;
    v4f f = *(const v4f*)(Qp + (size_t)(q0 + r) * DK_ + c4 * 4);
    v4bf h = { (__bf16)f.x, (__bf16)f.y, (__bf16)f.z, (__bf16)f.w };
    *(v4bf*)(sQw + r * PITCH + c4 * 4) = h;
  }
  const v16bf qa0 = loadAfrag(sQw + lm * PITCH +  0, hi);
  const v16bf qa1 = loadAfrag(sQw + lm * PITCH + 32, hi);

  float mrow[8], lrow[8];
  v8f acc[4];
  #pragma unroll
  for (int r = 0; r < 8; ++r) { mrow[r] = -1.0e30f; lrow[r] = 0.0f; }
  #pragma unroll
  for (int nb = 0; nb < 4; ++nb)
    #pragma unroll
    for (int i = 0; i < 8; ++i) acc[nb][i] = 0.0f;

  if (wave == 0) __builtin_amdgcn_s_wait_tensorcnt(0);
  __syncthreads();

  // mask folded into uniform scalars: s = c*sc + sb
  const float sc = masked ? 0.0f : 0.125f;     // 1/sqrt(64) or kill
  const float sb = masked ? -1.0e6f : 0.0f;
  const float LOG2E = 1.44269504088896f;
  __bf16* sPw = sQw;                            // reuse wave-private slice for P

  for (int t = 0; t < NT; ++t) {
    const int cur = t & 1;
    if (wave == 0 && (t + 1) < NT) {            // DMA next tile into other buffer
      const int nxt = (t + 1) & 1;
      tdm_load_tile_2d(lds_offset(&sK[nxt][0]),  Kp + (size_t)(t + 1) * KTILE * DK_,
                       64, 64, (unsigned long long)DK_);
      tdm_load_tile_2d(lds_offset(&sVt[nxt][0]), Vp + (size_t)(t + 1) * KTILE,
                       64, 64, (unsigned long long)S_);
    }

    // ---- scores: Q (16x64) x K^T -> 4 n-blocks, 8 WMMAs ----
    v8f c[4];
    #pragma unroll
    for (int nb = 0; nb < 4; ++nb) {
      const __bf16* krow = &sK[cur][(nb * 16 + lm) * PITCH];
      v16bf kb0 = loadBfrag(krow +  0, hi);
      v16bf kb1 = loadBfrag(krow + 32, hi);
      v8f z;
      #pragma unroll
      for (int i = 0; i < 8; ++i) z[i] = 0.0f;
      z = __builtin_amdgcn_wmma_f32_16x16x32_bf16(false, qa0, false, kb0, (short)0, z, false, false);
      z = __builtin_amdgcn_wmma_f32_16x16x32_bf16(false, qa1, false, kb1, (short)0, z, false, false);
      c[nb] = z;
    }

    // ---- scale + mask (uniform fma), online softmax ----
    #pragma unroll
    for (int nb = 0; nb < 4; ++nb)
      #pragma unroll
      for (int i = 0; i < 8; ++i)
        c[nb][i] = c[nb][i] * sc + sb;

    float alpha[8], mnew[8];
    #pragma unroll
    for (int r = 0; r < 8; ++r) {
      float tm = fmaxf(fmaxf(c[0][r], c[1][r]), fmaxf(c[2][r], c[3][r]));
      tm = redMax16(tm);
      mnew[r]  = fmaxf(mrow[r], tm);
      alpha[r] = __builtin_amdgcn_exp2f((mrow[r] - mnew[r]) * LOG2E);
      mrow[r]  = mnew[r];
    }
    #pragma unroll
    for (int nb = 0; nb < 4; ++nb)
      #pragma unroll
      for (int r = 0; r < 8; ++r)
        c[nb][r] = __builtin_amdgcn_exp2f((c[nb][r] - mnew[r]) * LOG2E);  // c := P
    #pragma unroll
    for (int r = 0; r < 8; ++r) {
      float ts = c[0][r] + c[1][r] + c[2][r] + c[3][r];
      ts = redSum16(ts);
      lrow[r] = lrow[r] * alpha[r] + ts;
      #pragma unroll
      for (int nb = 0; nb < 4; ++nb) acc[nb][r] *= alpha[r];
    }

    // ---- P (C-layout) -> LDS row-major -> A-fragments ----
    #pragma unroll
    for (int nb = 0; nb < 4; ++nb)
      #pragma unroll
      for (int r = 0; r < 8; ++r)
        sPw[(r + 8 * hi) * PITCH + nb * 16 + lm] = (__bf16)c[nb][r];
    v16bf pa0 = loadAfrag(sPw + lm * PITCH +  0, hi);
    v16bf pa1 = loadAfrag(sPw + lm * PITCH + 32, hi);

    // ---- O += P (16x64) x V (64x64) -> 8 WMMAs ----
    #pragma unroll
    for (int nb = 0; nb < 4; ++nb) {
      const __bf16* vrow = &sVt[cur][(nb * 16 + lm) * PITCH];
      v16bf vb0 = loadBfrag(vrow +  0, hi);
      v16bf vb1 = loadBfrag(vrow + 32, hi);
      acc[nb] = __builtin_amdgcn_wmma_f32_16x16x32_bf16(false, pa0, false, vb0, (short)0, acc[nb], false, false);
      acc[nb] = __builtin_amdgcn_wmma_f32_16x16x32_bf16(false, pa1, false, vb1, (short)0, acc[nb], false, false);
    }

    if (wave == 0 && (t + 1) < NT) __builtin_amdgcn_s_wait_tensorcnt(0);
    __syncthreads();
  }

  // ---- normalize and store (coalesced per C-VGPR) ----
  float inv[8];
  #pragma unroll
  for (int r = 0; r < 8; ++r) inv[r] = __builtin_amdgcn_rcpf(lrow[r]);
  #pragma unroll
  for (int nb = 0; nb < 4; ++nb)
    #pragma unroll
    for (int r = 0; r < 8; ++r) {
      int m = r + 8 * hi;
      Op[(size_t)(q0 + m) * DK_ + nb * 16 + lm] = acc[nb][r] * inv[r];
    }
}

// ============================ host launch ===================================

extern "C" void kernel_launch(void* const* d_in, const int* in_sizes, int n_in,
                              void* d_out, int out_size, void* d_ws, size_t ws_size,
                              hipStream_t stream) {
  (void)in_sizes; (void)n_in; (void)out_size; (void)ws_size;
  const float* q   = (const float*)d_in[0];
  const float* k   = (const float*)d_in[1];
  const float* v   = (const float*)d_in[2];
  const int*   msk = (const int*)d_in[3];
  float* out = (float*)d_out;

  const size_t elems = (size_t)BH_ * S_ * DK_;       // 4,194,304
  __bf16* kb = (__bf16*)d_ws;                        // 8 MB
  __bf16* vt = kb + elems;                           // 8 MB

  const int n4 = (int)(elems / 4);
  convert_k_kernel<<<dim3((n4 + 255) / 256), dim3(256), 0, stream>>>(k, kb, n4);
  transpose_v_kernel<<<dim3(S_ / 64, BH_), dim3(256), 0, stream>>>(v, vt);

  dim3 grid(S_ / QTILE, BH_);                        // 16 q-tiles x 32 heads
  flash_attn_cdna5_kernel<<<grid, dim3(256), 0, stream>>>(q, kb, vt, msk, out);
}
